// MT_GAT_topk_shareEn_multiple8_joint_last_70712341561397
// MI455X (gfx1250) — matrix operand order repaired and checked by reading, not compiled, not run on hardware
//
#include <hip/hip_runtime.h>
#include <hip/hip_bf16.h>
#include <math.h>

// ---------------- problem constants ----------------
#define N_NODES 33792      // B*S
#define CC      264        // channels
#define BB      128        // graphs
#define SS      264        // nodes per graph
#define HH      256        // GAT hidden
#define NHEADS  2
#define HDIM    132        // CC / NHEADS
#define EE      1081344    // B * EPG
#define KK      212        // top-k kept
#define FIN     54272      // KK * HH
#define BN_SCALE 0.9999950000374997f
#define ATTN_SCALE 0.08703882797784892f   // 1/sqrt(132)

typedef __attribute__((ext_vector_type(16))) __bf16 v16bf;
typedef __attribute__((ext_vector_type(8)))  float  v8f;

// ---------------- small helpers ----------------
__device__ inline void atomicMaxF(float* addr, float val) {
  if (val >= 0.f) {
    atomicMax(reinterpret_cast<int*>(addr), __float_as_int(val));
  } else {
    atomicMin(reinterpret_cast<unsigned int*>(addr),
              static_cast<unsigned int>(__float_as_int(val)));
  }
}

__global__ void k_zero(float* p, long n) {
  long i = (long)blockIdx.x * blockDim.x + threadIdx.x;
  if (i < n) p[i] = 0.f;
}
__global__ void k_fill(float* p, long n, float v) {
  long i = (long)blockIdx.x * blockDim.x + threadIdx.x;
  if (i < n) p[i] = v;
}
__global__ void k_add_inplace(float* a, const float* __restrict__ b, long n) {
  long i = (long)blockIdx.x * blockDim.x + threadIdx.x;
  if (i < n) a[i] += b[i];
}
__global__ void k_relu_ip(float* a, long n) {
  long i = (long)blockIdx.x * blockDim.x + threadIdx.x;
  if (i < n) a[i] = fmaxf(a[i], 0.f);
}
// split-K epilogue: D[i] = act(D[i] + bias[col])
__global__ void k_bias_relu(float* D, const float* __restrict__ bias, int Nr, long n, int relu) {
  long i = (long)blockIdx.x * blockDim.x + threadIdx.x;
  if (i >= n) return;
  float v = D[i] + (bias ? bias[(int)(i % Nr)] : 0.f);
  if (relu) v = fmaxf(v, 0.f);
  D[i] = v;
}

// ---------------- WMMA fragment loaders ----------------
// A fragment (16-bit A 16x32, this code's contiguous k order, not the ISA table's): lane L: row = tile_m + (L&15);
//   elem i (0..15) holds K = k0 + i + 16*(L>>4).
__device__ inline v16bf load_frag_a_vec(const float* __restrict__ row, int k0, int half) {
  const float4* p0 = reinterpret_cast<const float4*>(row + k0 + half * 16);
  const float4* p1 = reinterpret_cast<const float4*>(row + k0 + 8 + half * 16);
  float4 a0 = p0[0], a1 = p0[1], a2 = p1[0], a3 = p1[1];
  v16bf a;
  a[0]=(__bf16)a0.x;  a[1]=(__bf16)a0.y;  a[2]=(__bf16)a0.z;  a[3]=(__bf16)a0.w;
  a[4]=(__bf16)a1.x;  a[5]=(__bf16)a1.y;  a[6]=(__bf16)a1.z;  a[7]=(__bf16)a1.w;
  a[8]=(__bf16)a2.x;  a[9]=(__bf16)a2.y;  a[10]=(__bf16)a2.z; a[11]=(__bf16)a2.w;
  a[12]=(__bf16)a3.x; a[13]=(__bf16)a3.y; a[14]=(__bf16)a3.z; a[15]=(__bf16)a3.w;
  return a;
}
// Tail: branchless — clamp index to a valid address, load unconditionally, select 0.
__device__ inline v16bf load_frag_a_tail(const float* __restrict__ row, int k0, int kend,
                                         int half) {
  v16bf a;
#pragma unroll
  for (int i = 0; i < 16; ++i) {
    int ka = k0 + i + half * 16;
    int kc = (ka < kend) ? ka : (kend - 1);
    float v = row[kc];
    a[i] = (__bf16)((ka < kend) ? v : 0.f);
  }
  return a;
}
// B fragment (32x16): lane L: col = tile_n + (L&15); elem i holds K = k0 + i + 16*(L>>4).
// Row pointer is pre-clamped for out-of-range columns; garbage there only reaches
// accumulator lanes that are never stored (column n of D depends only on column n of B).
__device__ inline v16bf load_frag_b_vec(const float* __restrict__ row, int k0, int half) {
  const float4* p = reinterpret_cast<const float4*>(row + k0 + half * 16);
  float4 b0 = p[0], b1 = p[1], b2 = p[2], b3 = p[3];
  v16bf b;
  b[0]=(__bf16)b0.x;  b[1]=(__bf16)b0.y;  b[2]=(__bf16)b0.z;  b[3]=(__bf16)b0.w;
  b[4]=(__bf16)b1.x;  b[5]=(__bf16)b1.y;  b[6]=(__bf16)b1.z;  b[7]=(__bf16)b1.w;
  b[8]=(__bf16)b2.x;  b[9]=(__bf16)b2.y;  b[10]=(__bf16)b2.z; b[11]=(__bf16)b2.w;
  b[12]=(__bf16)b3.x; b[13]=(__bf16)b3.y; b[14]=(__bf16)b3.z; b[15]=(__bf16)b3.w;
  return b;
}
__device__ inline v16bf load_frag_b_tail(const float* __restrict__ row, int k0, int kend,
                                         int half) {
  v16bf b;
#pragma unroll
  for (int i = 0; i < 16; ++i) {
    int kb = k0 + i + 16 * half;
    int kc = (kb < kend) ? kb : (kend - 1);
    float v = row[kc];
    b[i] = (__bf16)((kb < kend) ? v : 0.f);
  }
  return b;
}

// ---------------- WMMA bf16 GEMM:  D[M,Nr] = A[M,Kr] @ W[Nr,Kr]^T (+bias)(+relu) ----------
// One wave32 per 64x32 output tile: 4 A-frags x 2 B-frags -> 8 wmma per K-step
// (3 b128 loads per wmma). M must be a multiple of 64; Nr, Kr arbitrary.
// launch_bounds(32,1): single-wave workgroup, optimize for 1 wave/EU so the
// ~150 live VGPRs (64 acc + 48 frag + addressing) fit without scratch spills.
// ksplit>1: grid.z K-chunks accumulate into pre-zeroed D via atomicAdd (no bias/relu).
// C/D layout: lane L: col = tile_n + (L&15); vgpr i: row = tile_m + i + 8*(L>>4).
__global__ __launch_bounds__(32, 1)
void k_gemm_wmma(const float* __restrict__ A, const float* __restrict__ W,
                 const float* __restrict__ bias, float* __restrict__ D,
                 int M, int Nr, int Kr, int relu, int ksplit) {
  const int lane = threadIdx.x & 31;
  const int m0 = blockIdx.x * 64;
  const int n0 = blockIdx.y * 32;
  const int r = lane & 15;
  const int half = lane >> 4;

  int kchunk = (Kr + ksplit - 1) / ksplit;
  kchunk = ((kchunk + 31) >> 5) << 5;            // 32-aligned chunks
  const int kbeg = blockIdx.z * kchunk;
  const int kend = min(Kr, kbeg + kchunk);

  const float* Arow0 = A + (size_t)(m0 + r) * (size_t)Kr;
  const float* Arow1 = A + (size_t)(m0 + 16 + r) * (size_t)Kr;
  const float* Arow2 = A + (size_t)(m0 + 32 + r) * (size_t)Kr;
  const float* Arow3 = A + (size_t)(m0 + 48 + r) * (size_t)Kr;
  const int nc0 = n0 + r;
  const int nc1 = n0 + 16 + r;
  const float* Wrow0 = W + (size_t)(nc0 < Nr ? nc0 : Nr - 1) * (size_t)Kr;
  const float* Wrow1 = W + (size_t)(nc1 < Nr ? nc1 : Nr - 1) * (size_t)Kr;

  v8f acc00 = {}, acc01 = {}, acc10 = {}, acc11 = {};
  v8f acc20 = {}, acc21 = {}, acc30 = {}, acc31 = {};

  auto mma_step = [&](v16bf a0, v16bf a1, v16bf a2, v16bf a3, v16bf b0, v16bf b1) {
    acc00 = __builtin_amdgcn_wmma_f32_16x16x32_bf16(false, a0, false, b0, (short)0, acc00, false, false);
    acc01 = __builtin_amdgcn_wmma_f32_16x16x32_bf16(false, a0, false, b1, (short)0, acc01, false, false);
    acc10 = __builtin_amdgcn_wmma_f32_16x16x32_bf16(false, a1, false, b0, (short)0, acc10, false, false);
    acc11 = __builtin_amdgcn_wmma_f32_16x16x32_bf16(false, a1, false, b1, (short)0, acc11, false, false);
    acc20 = __builtin_amdgcn_wmma_f32_16x16x32_bf16(false, a2, false, b0, (short)0, acc20, false, false);
    acc21 = __builtin_amdgcn_wmma_f32_16x16x32_bf16(false, a2, false, b1, (short)0, acc21, false, false);
    acc30 = __builtin_amdgcn_wmma_f32_16x16x32_bf16(false, a3, false, b0, (short)0, acc30, false, false);
    acc31 = __builtin_amdgcn_wmma_f32_16x16x32_bf16(false, a3, false, b1, (short)0, acc31, false, false);
  };

  // main loop: full 32-wide K-steps, fully vectorized b128 loads, no tail checks
  const bool vec_ok = ((Kr & 3) == 0);
  int kfull = vec_ok ? (kbeg + (((kend - kbeg) >> 5) << 5)) : kbeg;
  for (int k0 = kbeg; k0 < kfull; k0 += 32) {
    // stream-ahead prefetch (speculative; dropped silently past the end)
    __builtin_prefetch(Arow0 + k0 + 128, 0, 3);
    __builtin_prefetch(Arow1 + k0 + 128, 0, 3);
    __builtin_prefetch(Arow2 + k0 + 128, 0, 3);
    __builtin_prefetch(Arow3 + k0 + 128, 0, 3);
    __builtin_prefetch(Wrow0 + k0 + 128, 0, 3);
    __builtin_prefetch(Wrow1 + k0 + 128, 0, 3);
    mma_step(load_frag_a_vec(Arow0, k0, half),
             load_frag_a_vec(Arow1, k0, half),
             load_frag_a_vec(Arow2, k0, half),
             load_frag_a_vec(Arow3, k0, half),
             load_frag_b_vec(Wrow0, k0, half),
             load_frag_b_vec(Wrow1, k0, half));
  }
  // tail: <=1 iteration for Kr%32!=0 (or the whole range when Kr%4!=0)
  for (int k0 = kfull; k0 < kend; k0 += 32) {
    mma_step(load_frag_a_tail(Arow0, k0, kend, half),
             load_frag_a_tail(Arow1, k0, kend, half),
             load_frag_a_tail(Arow2, k0, kend, half),
             load_frag_a_tail(Arow3, k0, kend, half),
             load_frag_b_tail(Wrow0, k0, kend, half),
             load_frag_b_tail(Wrow1, k0, kend, half));
  }

  auto store_tile = [&](v8f acc, int mbase, int ncol) {
    if (ncol >= Nr) return;
    if (ksplit > 1) {
#pragma unroll
      for (int i = 0; i < 8; ++i)
        atomicAdd(&D[(size_t)(mbase + i + half * 8) * (size_t)Nr + ncol], acc[i]);
    } else {
      float bv = bias ? bias[ncol] : 0.f;
#pragma unroll
      for (int i = 0; i < 8; ++i) {
        float v = acc[i] + bv;
        if (relu) v = fmaxf(v, 0.f);
        D[(size_t)(mbase + i + half * 8) * (size_t)Nr + ncol] = v;
      }
    }
  };
  store_tile(acc00, m0,      nc0);
  store_tile(acc01, m0,      nc1);
  store_tile(acc10, m0 + 16, nc0);
  store_tile(acc11, m0 + 16, nc1);
  store_tile(acc20, m0 + 32, nc0);
  store_tile(acc21, m0 + 32, nc1);
  store_tile(acc30, m0 + 48, nc0);
  store_tile(acc31, m0 + 48, nc1);
}

static inline void gemm(const float* A, const float* W, const float* bias, float* D,
                        int M, int Nr, int Kr, int relu, hipStream_t st, int ksplit = 1) {
  dim3 g((unsigned)(M / 64), (unsigned)((Nr + 31) / 32), (unsigned)ksplit);
  k_gemm_wmma<<<g, 32, 0, st>>>(A, W, bias, D, M, Nr, Kr, relu, ksplit);
}

// ---------------- GINE edge aggregation ----------------
__global__ void k_edge_agg(const float* __restrict__ x, const int* __restrict__ src,
                           const int* __restrict__ dst, const float* __restrict__ ew,
                           const float* __restrict__ we, const float* __restrict__ be,
                           float* __restrict__ agg) {
  int e = blockIdx.x;
  int s = src[e], d = dst[e];
  float w = ew[e];
  const float* xr = x + (size_t)s * CC;
  float* ar = agg + (size_t)d * CC;
  for (int c = threadIdx.x; c < CC; c += blockDim.x) {
    float v = xr[c] + w * we[c] + be[c];
    v = fmaxf(v, 0.f);
    atomicAdd(&ar[c], v);
  }
}

// ---------------- BN / residual elementwise ----------------
__global__ void k_bn_res(float* hb, const float* __restrict__ x,
                         const float* __restrict__ g, const float* __restrict__ b, long n) {
  long i = (long)blockIdx.x * blockDim.x + threadIdx.x;
  if (i >= n) return;
  int c = (int)(i % CC);
  hb[i] = g[c] * ((hb[i] + x[i]) * BN_SCALE) + b[c];
}
// out = bn2(o + x) + h1
__global__ void k_out_combine(float* o, const float* __restrict__ x, const float* __restrict__ h1,
                              const float* __restrict__ g, const float* __restrict__ b, long n) {
  long i = (long)blockIdx.x * blockDim.x + threadIdx.x;
  if (i >= n) return;
  int c = (int)(i % CC);
  o[i] = g[c] * ((o[i] + x[i]) * BN_SCALE) + b[c] + h1[i];
}
// xg = relu(bn3(out + mlp))
__global__ void k_xg(float* o, const float* __restrict__ m,
                     const float* __restrict__ g, const float* __restrict__ b, long n) {
  long i = (long)blockIdx.x * blockDim.x + threadIdx.x;
  if (i >= n) return;
  int c = (int)(i % CC);
  o[i] = fmaxf(g[c] * ((o[i] + m[i]) * BN_SCALE) + b[c], 0.f);
}

// ---------------- dense attention (per graph, per head, per query row) ----------------
#define ATHR 128
__global__ void k_attn(const float* __restrict__ qkv, float* __restrict__ o) {
  __shared__ float p[SS];
  __shared__ float qv[HDIM];
  __shared__ float red[ATHR];
  int gid = blockIdx.x;
  int b = gid / (NHEADS * SS);
  int rem = gid % (NHEADS * SS);
  int h = rem / SS;
  int qr = rem % SS;
  int t = threadIdx.x;
  size_t rowbase = ((size_t)b * SS + qr) * (size_t)(3 * CC);
  for (int d = t; d < HDIM; d += ATHR) qv[d] = qkv[rowbase + h * HDIM + d];
  __syncthreads();
  float lmax = -3.0e38f;
  for (int kk = t; kk < SS; kk += ATHR) {
    const float* krow = qkv + ((size_t)b * SS + kk) * (size_t)(3 * CC) + CC + h * HDIM;
    float acc = 0.f;
    for (int d = 0; d < HDIM; ++d) acc += qv[d] * krow[d];
    acc *= ATTN_SCALE;
    p[kk] = acc;
    lmax = fmaxf(lmax, acc);
  }
  red[t] = lmax; __syncthreads();
  for (int s = ATHR / 2; s > 0; s >>= 1) { if (t < s) red[t] = fmaxf(red[t], red[t + s]); __syncthreads(); }
  float m = red[0]; __syncthreads();
  float lsum = 0.f;
  for (int kk = t; kk < SS; kk += ATHR) { float e = expf(p[kk] - m); p[kk] = e; lsum += e; }
  red[t] = lsum; __syncthreads();
  for (int s = ATHR / 2; s > 0; s >>= 1) { if (t < s) red[t] += red[t + s]; __syncthreads(); }
  float inv = 1.f / red[0];
  __syncthreads();
  for (int d = t; d < HDIM; d += ATHR) {
    float acc = 0.f;
    for (int kk = 0; kk < SS; ++kk)
      acc += p[kk] * qkv[((size_t)b * SS + kk) * (size_t)(3 * CC) + 2 * CC + h * HDIM + d];
    o[((size_t)b * SS + qr) * (size_t)CC + h * HDIM + d] = acc * inv;
  }
}

// ---------------- GAT ----------------
__global__ void k_nodesum(const float* __restrict__ xg, const float* __restrict__ as,
                          const float* __restrict__ ad, float* s, float* d) {
  int n = blockIdx.x * blockDim.x + threadIdx.x;
  if (n >= N_NODES) return;
  const float* r = xg + (size_t)n * HH;
  float ss = 0.f, dd = 0.f;
  for (int c = 0; c < HH; ++c) { float v = r[c]; ss += v * as[c]; dd += v * ad[c]; }
  s[n] = ss; d[n] = dd;
}
__global__ void k_alpha_max(const int* __restrict__ src, const int* __restrict__ dst,
                            const float* __restrict__ asum, const float* __restrict__ adsum,
                            float* __restrict__ ealpha, float* __restrict__ nmax) {
  long e = (long)blockIdx.x * blockDim.x + threadIdx.x;
  if (e >= (long)EE + N_NODES) return;
  int s, d;
  if (e < EE) { s = src[e]; d = dst[e]; } else { s = d = (int)(e - EE); }
  float a = asum[s] + adsum[d];
  a = (a > 0.f) ? a : 0.2f * a;   // leaky_relu 0.2
  ealpha[e] = a;
  atomicMaxF(&nmax[d], a);
}
__global__ void k_alpha_exp(const int* __restrict__ src, const int* __restrict__ dst,
                            float* __restrict__ ealpha, const float* __restrict__ nmax,
                            float* __restrict__ nsum) {
  long e = (long)blockIdx.x * blockDim.x + threadIdx.x;
  if (e >= (long)EE + N_NODES) return;
  int d = (e < EE) ? dst[e] : (int)(e - EE);
  float ex = expf(ealpha[e] - nmax[d]);
  ealpha[e] = ex;
  atomicAdd(&nsum[d], ex);
}
__global__ void k_gat_scatter(const int* __restrict__ src, const int* __restrict__ dst,
                              const float* __restrict__ ealpha, const float* __restrict__ nsum,
                              const float* __restrict__ xg, float* __restrict__ outb) {
  long e = blockIdx.x;
  int s, d;
  if (e < EE) { s = src[e]; d = dst[e]; } else { s = d = (int)(e - EE); }
  float w = ealpha[e] / nsum[d];
  int c = threadIdx.x;   // HH threads
  atomicAdd(&outb[(size_t)d * HH + c], w * xg[(size_t)s * HH + c]);
}
__global__ void k_gat_final(float* xb, const float* __restrict__ bias, long n) {
  long i = (long)blockIdx.x * blockDim.x + threadIdx.x;
  if (i >= n) return;
  xb[i] = fmaxf(xb[i] + bias[i % HH], 0.f);
}

// ---------------- node-attention gate ----------------
__global__ void k_conv4(const float* __restrict__ x11, const float* __restrict__ x22,
                        const float* __restrict__ w, const float* __restrict__ b,
                        float* __restrict__ sc) {
  int n = blockIdx.x * blockDim.x + threadIdx.x;
  if (n >= N_NODES) return;
  const float* r1 = x11 + (size_t)n * HH;
  const float* r2 = x22 + (size_t)n * HH;
  float acc = 0.f;
  for (int c = 0; c < HH; ++c) acc += r1[c] * w[c];
  for (int c = 0; c < HH; ++c) acc += r2[c] * w[HH + c];
  sc[n] = acc + b[0];
}
__global__ void k_att_sig(float* attbuf, float* o_att, long n) {
  long i = (long)blockIdx.x * blockDim.x + threadIdx.x;
  if (i >= n) return;
  float v = 1.f / (1.f + expf(-attbuf[i]));
  attbuf[i] = v;
  o_att[i] = v;
}
__global__ void k_cross(const float* __restrict__ x11, const float* __restrict__ x22,
                        const float* __restrict__ att, float* __restrict__ xc1,
                        float* __restrict__ xc2) {
  long n = blockIdx.x;
  int c = threadIdx.x;   // HH threads
  float a = att[n];
  float v1 = x11[(size_t)n * HH + c];
  float v2 = x22[(size_t)n * HH + c];
  xc1[(size_t)n * HH + c] = v1 + a * v2;
  xc2[(size_t)n * HH + c] = v2 + a * v1;
}

// ---------------- top-k pooling ----------------
__global__ void k_norm(const float* __restrict__ w1, const float* __restrict__ w2, float* out) {
  __shared__ float red[HH];
  const float* w = (blockIdx.x == 0) ? w1 : w2;
  float v = w[threadIdx.x];
  red[threadIdx.x] = v * v; __syncthreads();
  for (int s = HH / 2; s > 0; s >>= 1) { if (threadIdx.x < s) red[threadIdx.x] += red[threadIdx.x + s]; __syncthreads(); }
  if (threadIdx.x == 0) out[blockIdx.x] = sqrtf(red[0]);
}
__global__ void k_score(const float* __restrict__ xc, const float* __restrict__ w,
                        const float* __restrict__ normp, int idx, float* __restrict__ score) {
  int n = blockIdx.x * blockDim.x + threadIdx.x;
  if (n >= N_NODES) return;
  float inv = 1.f / normp[idx];
  const float* r = xc + (size_t)n * HH;
  float acc = 0.f;
  for (int c = 0; c < HH; ++c) acc += r[c] * w[c];
  score[n] = 1.f / (1.f + expf(-acc * inv));
}
// per-graph selection: descending values, ties -> smaller index first (matches lax.top_k)
__global__ void k_topk(const float* __restrict__ score, const float* __restrict__ xc,
                       float* __restrict__ pool, float* __restrict__ permOut,
                       float* __restrict__ scOut) {
  __shared__ float sc[SS];
  int b = blockIdx.x;
  int t = threadIdx.x;
  for (int i = t; i < SS; i += blockDim.x) sc[i] = score[(size_t)b * SS + i];
  __syncthreads();
  for (int i = t; i < SS; i += blockDim.x) {
    float si = sc[i];
    int rank = 0;
    for (int j = 0; j < SS; ++j) {
      float sj = sc[j];
      rank += (sj > si) || (sj == si && j < i);
    }
    if (rank < KK) {
      long node = (long)b * SS + i;
      long pos = (long)b * KK + rank;
      permOut[pos] = (float)node;
      scOut[pos] = 1.f / (1.f + expf(-si));   // sigmoid applied again, as in reference
      const float* xr = xc + (size_t)node * HH;
      float* pr = pool + (size_t)pos * HH;
      for (int c = 0; c < HH; ++c) pr[c] = xr[c] * si;
    }
  }
}

// ---------------- tiny heads ----------------
__global__ void k_y1(const float* __restrict__ hid, const float* __restrict__ w,
                     const float* __restrict__ b, float* __restrict__ out) {
  int r = blockIdx.x * blockDim.x + threadIdx.x;
  if (r >= BB) return;
  const float* hr = hid + (size_t)r * 512;
  float acc = 0.f;
  for (int c = 0; c < 512; ++c) acc += hr[c] * w[c];
  out[r] = acc + b[0];
}
__global__ void k_y2(const float* __restrict__ hid, const float* __restrict__ w,
                     const float* __restrict__ b, float* __restrict__ out) {
  int r = blockIdx.x * blockDim.x + threadIdx.x;
  if (r >= BB) return;
  const float* hr = hid + (size_t)r * 512;
  float a0 = b[0], a1 = b[1];
  for (int c = 0; c < 512; ++c) { float h = hr[c]; a0 += h * w[c]; a1 += h * w[512 + c]; }
  float m = fmaxf(a0, a1);
  float e0 = expf(a0 - m), e1 = expf(a1 - m);
  float inv = 1.f / (e0 + e1);
  out[(size_t)r * 2 + 0] = e0 * inv;
  out[(size_t)r * 2 + 1] = e1 * inv;
}

// ---------------- orchestration ----------------
extern "C" void kernel_launch(void* const* d_in, const int* in_sizes, int n_in,
                              void* d_out, int out_size, void* d_ws, size_t ws_size,
                              hipStream_t stream) {
  const float* x        = (const float*)d_in[0];
  const int*   ei       = (const int*)d_in[1];
  const float* ew       = (const float*)d_in[2];
  const float* gine_we  = (const float*)d_in[4];
  const float* gine_be  = (const float*)d_in[5];
  const float* gine_w1  = (const float*)d_in[6];
  const float* gine_b1  = (const float*)d_in[7];
  const float* gine_w2  = (const float*)d_in[8];
  const float* gine_b2  = (const float*)d_in[9];
  const float* wqkv     = (const float*)d_in[10];
  const float* bqkv     = (const float*)d_in[11];
  const float* wo       = (const float*)d_in[12];
  const float* bo       = (const float*)d_in[13];
  const float* mlp_w1   = (const float*)d_in[14];
  const float* mlp_b1   = (const float*)d_in[15];
  const float* mlp_w2   = (const float*)d_in[16];
  const float* mlp_b2   = (const float*)d_in[17];
  const float* bn1_g    = (const float*)d_in[18];
  const float* bn1_b    = (const float*)d_in[19];
  const float* bn2_g    = (const float*)d_in[20];
  const float* bn2_b    = (const float*)d_in[21];
  const float* bn3_g    = (const float*)d_in[22];
  const float* bn3_b    = (const float*)d_in[23];
  const float* gat1_w   = (const float*)d_in[24];
  const float* gat1_as  = (const float*)d_in[25];
  const float* gat1_ad  = (const float*)d_in[26];
  const float* gat1_bs  = (const float*)d_in[27];
  const float* gat2_w   = (const float*)d_in[28];
  const float* gat2_as  = (const float*)d_in[29];
  const float* gat2_ad  = (const float*)d_in[30];
  const float* gat2_bs  = (const float*)d_in[31];
  const float* conv4_w  = (const float*)d_in[32];
  const float* conv4_b  = (const float*)d_in[33];
  const float* lin4_w   = (const float*)d_in[34];
  const float* lin4_b   = (const float*)d_in[35];
  const float* lin5_w   = (const float*)d_in[36];
  const float* lin5_b   = (const float*)d_in[37];
  const float* pool1_w  = (const float*)d_in[38];
  const float* pool2_w  = (const float*)d_in[39];
  const float* lin1_w   = (const float*)d_in[40];
  const float* lin1_b   = (const float*)d_in[41];
  const float* lin3_w   = (const float*)d_in[42];
  const float* lin3_b   = (const float*)d_in[43];
  const float* lin12_w  = (const float*)d_in[44];
  const float* lin12_b  = (const float*)d_in[45];
  const float* lin32_w  = (const float*)d_in[46];
  const float* lin32_b  = (const float*)d_in[47];

  float* out = (float*)d_out;

  // ---- workspace layout (floats) ----
  float* ws = (float*)d_ws;
  size_t cur = 0;
  auto alloc = [&](size_t n) { float* p = ws + cur; cur += n; return p; };
  float* bufA   = alloc((size_t)N_NODES * CC);          // z / out / pool1
  float* bufB   = alloc((size_t)N_NODES * CC);          // h1 / mlp-out / xc1
  float* bufMid = alloc((size_t)N_NODES * (2 * CC));    // gine-mid / attn-o / mlp-mid / x22,xc2
  float* bufQKV = alloc((size_t)N_NODES * (3 * CC));    // qkv / xg1,xg2,x11 / pool2
  float* asum   = alloc(N_NODES);
  float* adsum  = alloc(N_NODES);
  float* nmax   = alloc(N_NODES);
  float* nsum   = alloc(N_NODES);
  float* score  = alloc(N_NODES);
  float* scN    = alloc(N_NODES);
  float* attbuf = alloc(N_NODES);
  float* ealpha = alloc((size_t)EE + N_NODES);
  float* atth   = alloc((size_t)BB * 66);
  float* hid1   = alloc((size_t)BB * 512);
  float* hid2   = alloc((size_t)BB * 512);
  float* normbuf= alloc(2);
  (void)ws_size; (void)in_sizes; (void)n_in; (void)out_size;

  // aliases (lifetime-checked reuse)
  float* xg1 = bufQKV;
  float* xg2 = bufQKV + (size_t)N_NODES * HH;
  float* x11 = bufQKV + 2 * (size_t)N_NODES * HH;
  float* x22 = bufMid;
  float* xc1 = bufB;
  float* xc2 = bufMid + (size_t)N_NODES * HH;
  float* pool1 = bufA;
  float* pool2 = bufQKV;

  // output sections (flat concat in return order; ints emitted as float values)
  float* o_y1   = out;                  // 128
  float* o_perm1= out + 128;            // B*K
  float* o_sc1  = o_perm1 + (size_t)BB * KK;
  float* o_y2   = o_sc1  + (size_t)BB * KK;   // 256
  float* o_perm2= o_y2   + 2 * BB;
  float* o_sc2  = o_perm2 + (size_t)BB * KK;
  float* o_att  = o_sc2  + (size_t)BB * KK;   // N

  auto gs = [](long n, int b) { return (unsigned)((n + b - 1) / b); };
  const long nc = (long)N_NODES * CC;
  const long nh = (long)N_NODES * HH;
  const long etot = (long)EE + N_NODES;
  const long pf = (long)BB * KK * HH;
  const long hidn = (long)BB * 512;
  const int* esrc = ei;
  const int* edst = ei + EE;

  // ---- GPSConv: GINE local branch ----
  k_zero<<<gs(nc, 256), 256, 0, stream>>>(bufA, nc);
  k_edge_agg<<<EE, 256, 0, stream>>>(x, esrc, edst, ew, gine_we, gine_be, bufA);
  k_add_inplace<<<gs(nc, 256), 256, 0, stream>>>(bufA, x, nc);          // z = x + agg
  gemm(bufA, gine_w1, gine_b1, bufMid, N_NODES, CC, CC, 1, stream);
  gemm(bufMid, gine_w2, gine_b2, bufB, N_NODES, CC, CC, 0, stream);
  k_bn_res<<<gs(nc, 256), 256, 0, stream>>>(bufB, x, bn1_g, bn1_b, nc); // h1

  // ---- GPSConv: global attention branch ----
  gemm(x, wqkv, bqkv, bufQKV, N_NODES, 3 * CC, CC, 0, stream);
  k_attn<<<BB * NHEADS * SS, ATHR, 0, stream>>>(bufQKV, bufMid);        // o -> bufMid
  gemm(bufMid, wo, bo, bufA, N_NODES, CC, CC, 0, stream);
  k_out_combine<<<gs(nc, 256), 256, 0, stream>>>(bufA, x, bufB, bn2_g, bn2_b, nc); // out = h1+h2

  // ---- GPSConv MLP + BN3 ----
  gemm(bufA, mlp_w1, mlp_b1, bufMid, N_NODES, 2 * CC, CC, 1, stream);
  gemm(bufMid, mlp_w2, mlp_b2, bufB, N_NODES, CC, 2 * CC, 0, stream);
  k_xg<<<gs(nc, 256), 256, 0, stream>>>(bufA, bufB, bn3_g, bn3_b, nc);  // xg = bufA

  // ---- GAT projections (bufA still = xg) ----
  gemm(bufA, gat1_w, nullptr, xg1, N_NODES, HH, CC, 0, stream);
  gemm(bufA, gat2_w, nullptr, xg2, N_NODES, HH, CC, 0, stream);

  // ---- GAT branch 1 ----
  k_nodesum<<<gs(N_NODES, 256), 256, 0, stream>>>(xg1, gat1_as, gat1_ad, asum, adsum);
  k_fill<<<gs(N_NODES, 256), 256, 0, stream>>>(nmax, N_NODES, -3.0e38f);
  k_zero<<<gs(N_NODES, 256), 256, 0, stream>>>(nsum, N_NODES);
  k_zero<<<gs(nh, 256), 256, 0, stream>>>(x11, nh);
  k_alpha_max<<<gs(etot, 256), 256, 0, stream>>>(esrc, edst, asum, adsum, ealpha, nmax);
  k_alpha_exp<<<gs(etot, 256), 256, 0, stream>>>(esrc, edst, ealpha, nmax, nsum);
  k_gat_scatter<<<(unsigned)etot, HH, 0, stream>>>(esrc, edst, ealpha, nsum, xg1, x11);
  k_gat_final<<<gs(nh, 256), 256, 0, stream>>>(x11, gat1_bs, nh);

  // ---- GAT branch 2 ----
  k_nodesum<<<gs(N_NODES, 256), 256, 0, stream>>>(xg2, gat2_as, gat2_ad, asum, adsum);
  k_fill<<<gs(N_NODES, 256), 256, 0, stream>>>(nmax, N_NODES, -3.0e38f);
  k_zero<<<gs(N_NODES, 256), 256, 0, stream>>>(nsum, N_NODES);
  k_zero<<<gs(nh, 256), 256, 0, stream>>>(x22, nh);
  k_alpha_max<<<gs(etot, 256), 256, 0, stream>>>(esrc, edst, asum, adsum, ealpha, nmax);
  k_alpha_exp<<<gs(etot, 256), 256, 0, stream>>>(esrc, edst, ealpha, nmax, nsum);
  k_gat_scatter<<<(unsigned)etot, HH, 0, stream>>>(esrc, edst, ealpha, nsum, xg2, x22);
  k_gat_final<<<gs(nh, 256), 256, 0, stream>>>(x22, gat2_bs, nh);

  // ---- node attention gate ----
  k_conv4<<<gs(N_NODES, 256), 256, 0, stream>>>(x11, x22, conv4_w, conv4_b, scN);
  gemm(scN, lin4_w, lin4_b, atth, BB, 66, CC, 1, stream);
  gemm(atth, lin5_w, lin5_b, attbuf, BB, CC, 66, 0, stream);
  k_att_sig<<<gs(N_NODES, 256), 256, 0, stream>>>(attbuf, o_att, N_NODES);
  k_cross<<<N_NODES, HH, 0, stream>>>(x11, x22, attbuf, xc1, xc2);

  // ---- top-k pooling + heads ----
  k_norm<<<2, HH, 0, stream>>>(pool1_w, pool2_w, normbuf);

  k_score<<<gs(N_NODES, 256), 256, 0, stream>>>(xc1, pool1_w, normbuf, 0, score);
  k_topk<<<BB, 256, 0, stream>>>(score, xc1, pool1, o_perm1, o_sc1);
  k_relu_ip<<<gs(pf, 256), 256, 0, stream>>>(pool1, pf);
  k_zero<<<gs(hidn, 256), 256, 0, stream>>>(hid1, hidn);
  gemm(pool1, lin1_w, nullptr, hid1, BB, 512, FIN, 0, stream, 16);     // split-K
  k_bias_relu<<<gs(hidn, 256), 256, 0, stream>>>(hid1, lin1_b, 512, hidn, 1);
  k_y1<<<1, 128, 0, stream>>>(hid1, lin3_w, lin3_b, o_y1);

  k_score<<<gs(N_NODES, 256), 256, 0, stream>>>(xc2, pool2_w, normbuf, 1, score);
  k_topk<<<BB, 256, 0, stream>>>(score, xc2, pool2, o_perm2, o_sc2);
  k_relu_ip<<<gs(pf, 256), 256, 0, stream>>>(pool2, pf);
  k_zero<<<gs(hidn, 256), 256, 0, stream>>>(hid2, hidn);
  gemm(pool2, lin12_w, nullptr, hid2, BB, 512, FIN, 0, stream, 16);    // split-K
  k_bias_relu<<<gs(hidn, 256), 256, 0, stream>>>(hid2, lin12_b, 512, hidn, 1);
  k_y2<<<1, 128, 0, stream>>>(hid2, lin32_w, lin32_b, o_y2);
}
